// RobertaSelfAttention_89223650607453
// MI455X (gfx1250) — compile-verified, matrix operand order checked by reading, not run on hardware
//
#include <hip/hip_runtime.h>

#define B_   8
#define S_   1024
#define HID_ 1024
#define H_   16
#define D_   64
#define TROWS 2048   // dist_table padded to 2048 rows (row 2047 = zeros)

typedef __attribute__((ext_vector_type(16))) __bf16 v16bf;
typedef __attribute__((ext_vector_type(8)))  float  v8f;

__device__ __forceinline__ unsigned short f2bf(float f) {
    unsigned int u = __builtin_bit_cast(unsigned int, f);
    u += 0x7FFFu + ((u >> 16) & 1u);            // round-to-nearest-even
    return (unsigned short)(u >> 16);
}

// 16x32 bf16 fragment loader (A layout; B mirrors A when loaded from the
// transposed operand). Source is row-major u16 with leading dim `ld`.
__device__ __forceinline__ v16bf load_frag(const unsigned short* base, int ld,
                                           int k0, int rowoff) {
    int lane = threadIdx.x & 31;
    int r  = (lane & 15) + rowoff;
    int kb = (lane >> 4) * 8;
    const unsigned short* p = base + r * ld + k0 + kb;
    union { v16bf v; unsigned int u[8]; } f;
#pragma unroll
    for (int i = 0; i < 4; ++i) {
        f.u[i]     = *(const unsigned int*)(p + 2 * i);
        f.u[4 + i] = *(const unsigned int*)(p + 16 + 2 * i);
    }
    return f.v;
}

__device__ __forceinline__ v8f wmma_bf(v16bf a, v16bf b, v8f c) {
    return __builtin_amdgcn_wmma_f32_16x16x32_bf16(
        false, a, false, b, (short)0, c, false, false);
}

// async copy of `nchunk16` 16-byte chunks (one chunk per lane per op),
// contiguous in both global memory and LDS. Tracked by ASYNCcnt.
__device__ __forceinline__ void async_copy(void* ldsdst, const void* gsrc,
                                           int nchunk16 /* multiple of 32 */) {
    int lane = threadIdx.x & 31;
    unsigned int       lbase = (unsigned int)(uintptr_t)ldsdst;
    unsigned long long gbase = (unsigned long long)(uintptr_t)gsrc;
#pragma unroll
    for (int i = 0; i < nchunk16 / 32; ++i) {
        unsigned int       l = lbase + (unsigned)(i * 32 + lane) * 16u;
        unsigned long long g = gbase + (unsigned long long)(i * 32 + lane) * 16ull;
        asm volatile("global_load_async_to_lds_b128 %0, %1, off"
                     :: "v"(l), "v"(g) : "memory");
    }
}

// ---------------- conversion kernels ----------------
__global__ void k_cvt(const float* __restrict__ x, unsigned short* __restrict__ y, int n) {
    for (int i = blockIdx.x * blockDim.x + threadIdx.x; i < n;
         i += gridDim.x * blockDim.x)
        y[i] = f2bf(x[i]);
}

__global__ void k_cvt_table(const float* __restrict__ t, unsigned short* __restrict__ y) {
    const int n = TROWS * D_, valid = (2 * 1024 - 1) * D_;
    for (int i = blockIdx.x * blockDim.x + threadIdx.x; i < n;
         i += gridDim.x * blockDim.x)
        y[i] = (i < valid) ? f2bf(t[i]) : (unsigned short)0;
}

__global__ void k_cvt_wt(const float* __restrict__ w, unsigned short* __restrict__ wt) {
    for (int i = blockIdx.x * blockDim.x + threadIdx.x; i < HID_ * HID_;
         i += gridDim.x * blockDim.x) {
        int r = i >> 10, c = i & 1023;
        wt[c * HID_ + r] = f2bf(w[i]);           // transpose: [out][in]
    }
}

// ---------------- projection GEMM: X = HS @ W + b ----------------
// outH: bf16 [B,H,S,D]; outT (optional): bf16 [B,H,D,S]  (V == K bug path)
__global__ __launch_bounds__(256) void k_proj(
    const unsigned short* __restrict__ A,   // HS bf16 [B*S, HID]
    const unsigned short* __restrict__ Wt,  // W^T bf16 [HID(out), HID(in)]
    const float* __restrict__ bias,
    unsigned short* __restrict__ outH,
    unsigned short* __restrict__ outT) {
    __shared__ __align__(16) unsigned short As[16 * HID_];   // 32 KB m-strip

    int wave = threadIdx.x >> 5, lane = threadIdx.x & 31;
    // all 8 jobs of a block share one m-tile (64 n-jobs per m-tile, 8 | 64)
    int m0 = ((blockIdx.x * 8) >> 6) * 16;
    int n0 = ((blockIdx.x * 8 + wave) & 63) * 16;

    // stage A strip (16 x 1024 bf16 = 32 KB) to LDS with async copies:
    // 2048 chunks of 16B; 256 threads x 8 iterations
    {
        unsigned int       lb = (unsigned int)(uintptr_t)As;
        unsigned long long gb = (unsigned long long)(uintptr_t)(A + (size_t)m0 * HID_);
#pragma unroll
        for (int t = 0; t < 8; ++t) {
            unsigned int       l = lb + (unsigned)(t * 256 + threadIdx.x) * 16u;
            unsigned long long g = gb + (unsigned long long)(t * 256 + threadIdx.x) * 16ull;
            asm volatile("global_load_async_to_lds_b128 %0, %1, off"
                         :: "v"(l), "v"(g) : "memory");
        }
    }
    asm volatile("s_wait_asynccnt 0" ::: "memory");
    __syncthreads();

    v8f c = {};
    for (int k0 = 0; k0 < HID_; k0 += 32) {
        v16bf a = load_frag(As, HID_, k0, 0);                 // ds_load frags
        v16bf b = load_frag(Wt + n0 * HID_, HID_, k0, 0);     // global frags
        c = wmma_bf(a, b, c);
    }
    int nl = lane & 15, half = lane >> 4;
    int cg = n0 + nl;
    float bv = bias[cg];
    int h = cg >> 6, d = cg & 63;
#pragma unroll
    for (int v = 0; v < 8; ++v) {
        int row = m0 + v + half * 8;
        int bb = row >> 10, sp = row & 1023;
        unsigned short val = f2bf(c[v] + bv);
        outH[(((bb * H_ + h) * S_ + sp) * D_) + d] = val;
        if (outT) outT[(((bb * H_ + h) * D_ + d) * S_) + sp] = val;
    }
}

// ---------------- fused attention ----------------
// one block = one (b,h) and a 32-row query strip
__global__ __launch_bounds__(256) void k_attn(
    const unsigned short* __restrict__ Q,   // [B*H, S, D] bf16
    const unsigned short* __restrict__ K,   // [B*H, S, D] bf16 (also V)
    const unsigned short* __restrict__ Kt,  // [B*H, D, S] bf16 (V^T)
    const unsigned short* __restrict__ T,   // [2048, 64]  bf16 band table
    float* __restrict__ out) {              // [B, S, HID] f32
    __shared__ float sc[32][1024];                    // 128 KB score strip
    __shared__ float g[8][16][32];                    //  16 KB band scratch
    __shared__ __align__(16) unsigned short kst[8][2][16 * D_];  // 32 KB K dbl-buf

    int wave = threadIdx.x >> 5, lane = threadIdx.x & 31;
    int bh = blockIdx.x >> 5;
    int l0 = (blockIdx.x & 31) * 32;
    int bb = bh >> 4, h = bh & 15;
    int half = lane >> 4, nl = lane & 15;

    // ---- phase 1: scores = (QK^T + rel_q + rel_k) / 8 ----
    int lt = wave & 1;                       // fixed per wave -> hoist Q frags
    const unsigned short* qbase = Q + ((size_t)bh * S_ + l0 + lt * 16) * D_;
    v16bf qa0 = load_frag(qbase, D_, 0, 0);
    v16bf qa1 = load_frag(qbase, D_, 32, 0);

    const unsigned short* Kbh = K + (size_t)bh * S_ * D_;
    // prime the async pipeline: K tile for first job (2 KB contiguous)
    async_copy(&kst[wave][0][0], Kbh + (size_t)((wave >> 1) * 16) * D_, 128);

    int cur = 0;
    for (int job = wave; job < 128; job += 8) {   // job&1 == wave&1 == lt
        int r0 = (job >> 1) * 16;
        int nxt = job + 8;
        if (nxt < 128) {                           // prefetch next K tile
            async_copy(&kst[wave][cur ^ 1][0],
                       Kbh + (size_t)((nxt >> 1) * 16) * D_, 128);
            asm volatile("s_wait_asynccnt 4" ::: "memory");  // cur tile landed
        } else {
            asm volatile("s_wait_asynccnt 0" ::: "memory");
        }

        const unsigned short* kb = &kst[wave][cur][0];
        v16bf kf0 = load_frag(kb, D_, 0, 0);       // ds_load frags from LDS
        v16bf kf1 = load_frag(kb, D_, 32, 0);
        v8f cqk = {};
        cqk = wmma_bf(qa0, kf0, cqk);
        cqk = wmma_bf(qa1, kf1, cqk);

        // band of 31 table rows: dist = (l - r) + 1023, j = li - ri + 15
        int dbase = (l0 + lt * 16) - r0 + 1008;    // in [0, 2016]
        const unsigned short* tb = T + dbase * D_;
        v16bf t00 = load_frag(tb, D_, 0, 0);
        v16bf t01 = load_frag(tb, D_, 32, 0);
        v16bf t10 = load_frag(tb, D_, 0, 16);
        v16bf t11 = load_frag(tb, D_, 32, 16);

        v8f mlo = {}, mhi = {}, nlo = {}, nhi = {};
        mlo = wmma_bf(qa0, t00, mlo); mlo = wmma_bf(qa1, t01, mlo);
        mhi = wmma_bf(qa0, t10, mhi); mhi = wmma_bf(qa1, t11, mhi);
        nlo = wmma_bf(kf0, t00, nlo); nlo = wmma_bf(kf1, t01, nlo);
        nhi = wmma_bf(kf0, t10, nhi); nhi = wmma_bf(kf1, t11, nhi);

        // gather rel_q band via per-wave LDS scratch
        float acc[8];
#pragma unroll
        for (int v = 0; v < 8; ++v) {
            int m = v + half * 8;
            g[wave][m][nl]      = mlo[v];
            g[wave][m][16 + nl] = mhi[v];
        }
#pragma unroll
        for (int v = 0; v < 8; ++v) {
            int li = v + half * 8;
            acc[v] = cqk[v] + g[wave][li][li - nl + 15];
        }
        // reuse scratch for rel_k (DS ops are in-order per wave)
#pragma unroll
        for (int v = 0; v < 8; ++v) {
            int m = v + half * 8;
            g[wave][m][nl]      = nlo[v];
            g[wave][m][16 + nl] = nhi[v];
        }
#pragma unroll
        for (int v = 0; v < 8; ++v) {
            int li = v + half * 8;
            sc[lt * 16 + li][r0 + nl] =
                (acc[v] + g[wave][nl][li - nl + 15]) * 0.125f;
        }
        cur ^= 1;
    }
    __syncthreads();

    // ---- phase 2: row softmax, write probs as packed bf16 in place ----
    for (int rr = wave * 4; rr < wave * 4 + 4; ++rr) {
        float mx = -3.0e38f;
        for (int i = 0; i < 32; ++i) mx = fmaxf(mx, sc[rr][lane + 32 * i]);
        for (int off = 16; off > 0; off >>= 1) mx = fmaxf(mx, __shfl_xor(mx, off));
        float sum = 0.f;
        for (int i = 0; i < 32; ++i) sum += __expf(sc[rr][lane + 32 * i] - mx);
        for (int off = 16; off > 0; off >>= 1) sum += __shfl_xor(sum, off);
        float inv = 1.0f / sum;
        unsigned int* P32 = (unsigned int*)&sc[rr][0];
        for (int i = 0; i < 16; ++i) {         // write slot j after cols 2j,2j+1 read
            int j = lane + 32 * i;
            float a = sc[rr][2 * j], b2 = sc[rr][2 * j + 1];
            P32[j] = (unsigned int)f2bf(__expf(a - mx) * inv) |
                     ((unsigned int)f2bf(__expf(b2 - mx) * inv) << 16);
        }
    }
    __syncthreads();

    // ---- phase 3: ctx = P @ V (V == K), P read as bf16 from LDS ----
    int lt2 = wave & 1, dt = wave >> 1;        // 2 row-tiles x 4 d-tiles
    const unsigned short* Pu = (const unsigned short*)&sc[0][0] + lt2 * 16 * 2048;
    const unsigned short* vb = Kt + ((size_t)bh * D_ + dt * 16) * S_;
    v8f c = {};
    for (int k0 = 0; k0 < S_; k0 += 32) {
        v16bf a = load_frag(Pu, 2048, k0, 0);  // LDS ds_load fragments
        v16bf b = load_frag(vb, S_, k0, 0);
        c = wmma_bf(a, b, c);
    }
#pragma unroll
    for (int v = 0; v < 8; ++v) {
        int l = l0 + lt2 * 16 + v + half * 8;
        out[(bb * S_ + l) * HID_ + h * 64 + dt * 16 + nl] = c[v];
    }
}

extern "C" void kernel_launch(void* const* d_in, const int* in_sizes, int n_in,
                              void* d_out, int out_size, void* d_ws, size_t ws_size,
                              hipStream_t stream) {
    (void)in_sizes; (void)n_in; (void)out_size; (void)ws_size;
    const float* hs = (const float*)d_in[0];
    const float* Wq = (const float*)d_in[1];
    const float* bq = (const float*)d_in[2];
    const float* Wk = (const float*)d_in[3];
    const float* bk = (const float*)d_in[4];
    // d_in[5]/d_in[6] (Wv,bv) intentionally unused: reference V uses K projection
    const float* dt = (const float*)d_in[7];
    float* out = (float*)d_out;

    char* w = (char*)d_ws;                                   // ~68.3 MB total
    unsigned short* HSbf = (unsigned short*)w; w += (size_t)B_ * S_ * HID_ * 2;
    unsigned short* Wqt  = (unsigned short*)w; w += (size_t)HID_ * HID_ * 2;
    unsigned short* Wkt  = (unsigned short*)w; w += (size_t)HID_ * HID_ * 2;
    unsigned short* Tbf  = (unsigned short*)w; w += (size_t)TROWS * D_ * 2;
    unsigned short* Qbf  = (unsigned short*)w; w += (size_t)B_ * H_ * S_ * D_ * 2;
    unsigned short* Kbf  = (unsigned short*)w; w += (size_t)B_ * H_ * S_ * D_ * 2;
    unsigned short* Ktbf = (unsigned short*)w; w += (size_t)B_ * H_ * S_ * D_ * 2;

    k_cvt      <<<2048, 256, 0, stream>>>(hs, HSbf, B_ * S_ * HID_);
    k_cvt_table<<<256, 256, 0, stream>>>(dt, Tbf);
    k_cvt_wt   <<<1024, 256, 0, stream>>>(Wq, Wqt);
    k_cvt_wt   <<<1024, 256, 0, stream>>>(Wk, Wkt);

    k_proj<<<4096, 256, 0, stream>>>(HSbf, Wqt, bq, Qbf, (unsigned short*)nullptr);
    k_proj<<<4096, 256, 0, stream>>>(HSbf, Wkt, bk, Kbf, Ktbf);   // K and V^T

    k_attn<<<B_ * H_ * (S_ / 32), 256, 0, stream>>>(Qbf, Kbf, Ktbf, Tbf, out);
}
